// SoftTopKSAE_51977694216557
// MI455X (gfx1250) — matrix operand order repaired and checked by reading, not compiled, not run on hardware
//
#include <hip/hip_runtime.h>
#include <math.h>

typedef __attribute__((ext_vector_type(16))) __bf16 v16bf;
typedef __attribute__((ext_vector_type(8)))  __bf16 v8bf;
typedef __attribute__((ext_vector_type(4)))  __bf16 v4bf;
typedef __attribute__((ext_vector_type(8)))  float  v8f;

#define B_DIM 4096
#define D_DIM 2048
#define F_DIM 16384
#define CAP   256   // max kept entries per row (k_est <= 128)

// ---------------------------------------------------------------------------
// WMMA: D = A(16x32 bf16) x B(32x16 bf16) + C(16x16 f32), wave32, f32 accum.
// ---------------------------------------------------------------------------
__device__ __forceinline__ v8f wmma_bf16(v16bf a, v16bf b, v8f c) {
  return __builtin_amdgcn_wmma_f32_16x16x32_bf16(
      /*neg_a=*/false, a, /*neg_b=*/false, b,
      /*c_mod=*/(short)0, c, /*reuse_a=*/false, /*reuse_b=*/false);
}

// ---------------------------------------------------------------------------
// Elementwise converts
// ---------------------------------------------------------------------------
__global__ void cvt_center_bf16_kernel(const float* __restrict__ x,
                                       const float* __restrict__ bsub,
                                       __bf16* __restrict__ dst, size_t total) {
  size_t i = ((size_t)blockIdx.x * blockDim.x + threadIdx.x) * 4;
  if (i >= total) return;
  v4bf o;
#pragma unroll
  for (int t = 0; t < 4; ++t) {
    size_t j = i + t;
    o[t] = (__bf16)(x[j] - bsub[j & (D_DIM - 1)]);  // xc = x - b_dec
  }
  *(v4bf*)(dst + i) = o;
}

__global__ void cvt_bf16_kernel(const float* __restrict__ src,
                                __bf16* __restrict__ dst, size_t total) {
  size_t i = ((size_t)blockIdx.x * blockDim.x + threadIdx.x) * 4;
  if (i >= total) return;
  v4bf o;
#pragma unroll
  for (int t = 0; t < 4; ++t) o[t] = (__bf16)src[i + t];
  *(v4bf*)(dst + i) = o;
}

__global__ void zero_kernel(float* __restrict__ p, int n) {
  int i = blockIdx.x * blockDim.x + threadIdx.x;
  if (i < n) p[i] = 0.0f;
}

// ---------------------------------------------------------------------------
// bf16 WMMA GEMM:  v = relu( A[m,:] . W[n,:] + bias[n] )
// MODE 0: store v to Cout (encoder activations, f32)
// MODE 1: accumulate v * w2[n] over n into kdot[m] (k-estimator, no store)
//
// 256 threads = 8 waves arranged 2(M) x 4(N); each wave owns a 64x64 tile as
// a 4x4 grid of 16x16 fragments -> block tile 128 x 256.
// Per K=32 step: 8 A-loads(b128) + 8 B-loads(2xb128 pairs) feed 16 wmma.
//
// Fragment layouts (16-bit, this kernel's K order; ISA table order differs):
//   A 16x32: lane r=lane&15 -> M=r; half hh=lane>>4: vec elems e = A[M=r][K=16hh+e],
//            contiguous K (same order as B).
//   B 32x16: lane r -> N=r; half hh: vec elems e = B[K=16hh+e][N=r],
//            contiguous K -> single 32B chunk of W row.
//   C 16x16: vec elem e -> M = e + 8*hh, N = r.
// ---------------------------------------------------------------------------
template <int MODE>
__global__ __launch_bounds__(256) void gemm_bf16_kernel(
    const __bf16* __restrict__ A,    // [M,K] bf16 (xc)
    const __bf16* __restrict__ W,    // [N,K] bf16
    const float* __restrict__ bias,  // [N]
    const float* __restrict__ w2,    // [N]   (MODE 1)
    float* __restrict__ kdot,        // [M]   (MODE 1)
    float* __restrict__ Cout,        // [M,N] (MODE 0)
    int N_total, int K) {
  const int lane = threadIdx.x & 31;
  const int wave = threadIdx.x >> 5;
  const int wm = wave >> 2;  // 2 waves along M
  const int wn = wave & 3;   // 4 waves along N
  const int m0 = blockIdx.x * 128 + wm * 64;
  const int n0 = blockIdx.y * 256 + wn * 64;
  const int r = lane & 15;
  const int hh = lane >> 4;

  const __bf16* arow[4];
  const __bf16* wrow[4];
#pragma unroll
  for (int t = 0; t < 4; ++t) {
    arow[t] = A + (size_t)(m0 + 16 * t + r) * K;
    wrow[t] = W + (size_t)(n0 + 16 * t + r) * K;
  }

  v8f acc[4][4];
#pragma unroll
  for (int i = 0; i < 4; ++i)
#pragma unroll
    for (int j = 0; j < 4; ++j) acc[i][j] = (v8f){};

  for (int k = 0; k < K; k += 32) {
    v16bf af[4], bf[4];
#pragma unroll
    for (int mt = 0; mt < 4; ++mt) {
      af[mt] = *(const v16bf*)(arow[mt] + k + 16 * hh);
    }
#pragma unroll
    for (int nt = 0; nt < 4; ++nt)
      bf[nt] = *(const v16bf*)(wrow[nt] + k + 16 * hh);
#pragma unroll
    for (int mt = 0; mt < 4; ++mt)
#pragma unroll
      for (int nt = 0; nt < 4; ++nt)
        acc[mt][nt] = wmma_bf16(af[mt], bf[nt], acc[mt][nt]);
  }

  if (MODE == 0) {
#pragma unroll
    for (int mt = 0; mt < 4; ++mt)
#pragma unroll
      for (int nt = 0; nt < 4; ++nt) {
        const int n = n0 + 16 * nt + r;
        const float bv = bias[n];
#pragma unroll
        for (int e = 0; e < 8; ++e) {
          const int m = m0 + 16 * mt + 8 * hh + e;
          Cout[(size_t)m * N_total + n] = fmaxf(acc[mt][nt][e] + bv, 0.0f);
        }
      }
  } else {
    float w2v[4], bv[4];
#pragma unroll
    for (int nt = 0; nt < 4; ++nt) {
      const int n = n0 + 16 * nt + r;
      w2v[nt] = w2[n];
      bv[nt] = bias[n];
    }
#pragma unroll
    for (int mt = 0; mt < 4; ++mt)
#pragma unroll
      for (int e = 0; e < 8; ++e) {
        float s = 0.0f;
#pragma unroll
        for (int nt = 0; nt < 4; ++nt)
          s += fmaxf(acc[mt][nt][e] + bv[nt], 0.0f) * w2v[nt];
        // reduce across the 16 lanes (r = 0..15) sharing row m
        s += __shfl_xor(s, 1);
        s += __shfl_xor(s, 2);
        s += __shfl_xor(s, 4);
        s += __shfl_xor(s, 8);
        if (r == 0) atomicAdd(&kdot[m0 + 16 * mt + 8 * hh + e], s);
      }
  }
}

// ---------------------------------------------------------------------------
// k_est = sigmoid(kdot + b_kest2) * 2k
// ---------------------------------------------------------------------------
__global__ void kest_kernel(const float* __restrict__ kdot,
                            const float* __restrict__ b_kest2,
                            const int* __restrict__ kptr,
                            float* __restrict__ kest, int n) {
  int i = blockIdx.x * blockDim.x + threadIdx.x;
  if (i < n) {
    float z = kdot[i] + b_kest2[0];
    float s = 1.0f / (1.0f + expf(-z));
    kest[i] = s * 2.0f * (float)kptr[0];
  }
}

// ---------------------------------------------------------------------------
// Per-row exact top-k (radix select on float bits, values >= 0 post-ReLU).
// Emits compacted (index, value) list per row for the sparse decoder.
// ---------------------------------------------------------------------------
__global__ __launch_bounds__(256) void topk_select_kernel(
    const float* __restrict__ acts, const float* __restrict__ kest,
    int* __restrict__ idx, float* __restrict__ val, int* __restrict__ nnz,
    int F) {
  const int row = blockIdx.x;
  const int tid = threadIdx.x;
  const float* rowp = acts + (size_t)row * F;

  __shared__ int hist[256];
  __shared__ int sel_digit;
  __shared__ int sel_remaining;
  __shared__ int eq_taken;
  __shared__ int out_cnt;

  int kk = (int)ceilf(kest[row]);  // rank < k_est keeps ceil(k_est) elements
  if (kk < 0) kk = 0;
  if (kk > CAP) kk = CAP;  // k_est <= 2k = 128 < CAP

  unsigned prefix = 0;
  int remaining = kk;
  for (int shift = 24; shift >= 0; shift -= 8) {
    hist[tid] = 0;
    __syncthreads();
    const unsigned himask = (shift == 24) ? 0u : (0xFFFFFFFFu << (shift + 8));
    for (int j = tid; j < F; j += 256) {
      unsigned bits = __float_as_uint(rowp[j]);
      if ((bits & himask) == (prefix & himask))
        atomicAdd(&hist[(bits >> shift) & 255], 1);
    }
    __syncthreads();
    if (tid == 0) {
      int rem = remaining;
      int d = 255;
      for (; d > 0; --d) {
        if (hist[d] >= rem) break;
        rem -= hist[d];
      }
      sel_digit = d;
      sel_remaining = rem;
    }
    __syncthreads();
    prefix |= ((unsigned)sel_digit) << shift;
    remaining = sel_remaining;
    __syncthreads();
  }

  if (tid == 0) {
    eq_taken = 0;
    out_cnt = 0;
  }
  __syncthreads();
  const unsigned T = prefix;
  const int need_eq = remaining;
  for (int j = tid; j < F; j += 256) {
    float fv = rowp[j];
    unsigned bits = __float_as_uint(fv);
    bool keep;
    if (bits > T)
      keep = true;
    else if (bits == T)
      keep = (atomicAdd(&eq_taken, 1) < need_eq);
    else
      keep = false;
    if (keep && fv > 0.0f) {  // zero entries contribute nothing to the decode
      int p = atomicAdd(&out_cnt, 1);
      if (p < CAP) {
        idx[(size_t)row * CAP + p] = j;
        val[(size_t)row * CAP + p] = fv;
      }
    }
  }
  __syncthreads();
  if (tid == 0) nnz[row] = (out_cnt < CAP) ? out_cnt : CAP;
}

// ---------------------------------------------------------------------------
// W_dec [D,F] -> W_decT [F,D], fp32, LDS-tiled 32x32
// ---------------------------------------------------------------------------
__global__ __launch_bounds__(256) void transpose_kernel(
    const float* __restrict__ src, float* __restrict__ dst, int D, int F) {
  __shared__ float tile[32][33];
  const int tx = threadIdx.x & 31;
  const int ty = threadIdx.x >> 5;  // 0..7
  const int f0 = blockIdx.x * 32;
  const int d0 = blockIdx.y * 32;
#pragma unroll
  for (int i = 0; i < 4; ++i)
    tile[ty + 8 * i][tx] = src[(size_t)(d0 + ty + 8 * i) * F + (f0 + tx)];
  __syncthreads();
#pragma unroll
  for (int i = 0; i < 4; ++i)
    dst[(size_t)(f0 + ty + 8 * i) * D + (d0 + tx)] = tile[tx][ty + 8 * i];
}

// ---------------------------------------------------------------------------
// Sparse fp32 decoder: out[row,:] = sum_t val[t] * W_decT[idx[t],:] + b_dec
// One block per row; W_decT (128 MB) is L2-resident (192 MB L2).
// ---------------------------------------------------------------------------
__global__ __launch_bounds__(256) void decoder_sparse_kernel(
    const float* __restrict__ WdT, const float* __restrict__ b_dec,
    const int* __restrict__ idx, const float* __restrict__ val,
    const int* __restrict__ nnz, float* __restrict__ out) {
  const int row = blockIdx.x;
  const int tid = threadIdx.x;
  const int n = nnz[row];
  const int* ip = idx + (size_t)row * CAP;
  const float* vp = val + (size_t)row * CAP;

  const int c0 = tid * 4;         // columns [c0, c0+4)
  const int c1 = 1024 + tid * 4;  // columns [c1, c1+4)
  float a0 = 0.f, a1 = 0.f, a2 = 0.f, a3 = 0.f;
  float b0 = 0.f, b1 = 0.f, b2 = 0.f, b3 = 0.f;

  for (int t = 0; t < n; ++t) {
    const int j = ip[t];
    const float v = vp[t];
    const float* wr = WdT + (size_t)j * D_DIM;
    if (t + 1 < n)  // hint next gathered row toward L2/L0
      __builtin_prefetch(WdT + (size_t)ip[t + 1] * D_DIM + c0, 0, 0);
    float4 w0 = *(const float4*)(wr + c0);
    float4 w1 = *(const float4*)(wr + c1);
    a0 += v * w0.x; a1 += v * w0.y; a2 += v * w0.z; a3 += v * w0.w;
    b0 += v * w1.x; b1 += v * w1.y; b2 += v * w1.z; b3 += v * w1.w;
  }

  float4 bd0 = *(const float4*)(b_dec + c0);
  float4 bd1 = *(const float4*)(b_dec + c1);
  float* op = out + (size_t)row * D_DIM;
  float4 o0 = {a0 + bd0.x, a1 + bd0.y, a2 + bd0.z, a3 + bd0.w};
  float4 o1 = {b0 + bd1.x, b1 + bd1.y, b2 + bd1.z, b3 + bd1.w};
  *(float4*)(op + c0) = o0;
  *(float4*)(op + c1) = o1;
}

// ---------------------------------------------------------------------------
extern "C" void kernel_launch(void* const* d_in, const int* in_sizes, int n_in,
                              void* d_out, int out_size, void* d_ws,
                              size_t ws_size, hipStream_t stream) {
  (void)in_sizes; (void)n_in; (void)out_size; (void)ws_size;

  const float* x       = (const float*)d_in[0];
  const float* W_enc   = (const float*)d_in[1];
  const float* b_enc   = (const float*)d_in[2];
  const float* W_kest1 = (const float*)d_in[3];
  const float* b_kest1 = (const float*)d_in[4];
  const float* W_kest2 = (const float*)d_in[5];
  const float* b_kest2 = (const float*)d_in[6];
  const float* W_dec   = (const float*)d_in[7];
  const float* b_dec   = (const float*)d_in[8];
  const int*   kptr    = (const int*)d_in[9];

  // Workspace layout (bytes):
  char* ws = (char*)d_ws;
  float*  acts    = (float*)ws;                                   // 256 MiB
  char*   wregion = ws + (size_t)B_DIM * F_DIM * 4;               // 128 MiB
  __bf16* Wencb   = (__bf16*)wregion;                             //  64 MiB
  __bf16* Wk1b    = (__bf16*)(wregion + (size_t)F_DIM * D_DIM * 2);// 64 MiB
  float*  WdT     = (float*)wregion;       // reuses wregion after GEMMs
  char*   tail    = wregion + (size_t)F_DIM * D_DIM * 4;          // +128 MiB
  __bf16* xcb     = (__bf16*)tail;                                //  16 MiB
  char*   tail2   = tail + (size_t)B_DIM * D_DIM * 2;
  float*  kdot    = (float*)tail2;
  float*  kest    = kdot + B_DIM;
  int*    nnz     = (int*)(kest + B_DIM);
  int*    idx     = nnz + B_DIM;                                  //   4 MiB
  float*  val     = (float*)(idx + (size_t)B_DIM * CAP);          //   4 MiB
  float*  out     = (float*)d_out;

  const dim3 blk(256);

  zero_kernel<<<(B_DIM + 255) / 256, 256, 0, stream>>>(kdot, B_DIM);

  // bf16 conversions (x is centered by b_dec during conversion)
  {
    size_t txc = (size_t)B_DIM * D_DIM;
    cvt_center_bf16_kernel<<<(unsigned)(txc / 4 / 256), blk, 0, stream>>>(
        x, b_dec, xcb, txc);
    size_t tw = (size_t)F_DIM * D_DIM;
    cvt_bf16_kernel<<<(unsigned)(tw / 4 / 256), blk, 0, stream>>>(W_enc, Wencb, tw);
    cvt_bf16_kernel<<<(unsigned)(tw / 4 / 256), blk, 0, stream>>>(W_kest1, Wk1b, tw);
  }

  // Encoder: acts = relu(xc @ W_enc^T + b_enc)   [bf16 WMMA, f32 accum]
  const dim3 grid_enc(B_DIM / 128, F_DIM / 256);
  gemm_bf16_kernel<0><<<grid_enc, blk, 0, stream>>>(
      xcb, Wencb, b_enc, nullptr, nullptr, acts, F_DIM, D_DIM);

  // k-estimator: kdot[m] = relu(xc @ W_kest1^T + b_kest1) . W_kest2
  gemm_bf16_kernel<1><<<grid_enc, blk, 0, stream>>>(
      xcb, Wk1b, b_kest1, W_kest2, kdot, nullptr, F_DIM, D_DIM);

  kest_kernel<<<(B_DIM + 255) / 256, 256, 0, stream>>>(kdot, b_kest2, kptr,
                                                       kest, B_DIM);

  // Exact per-row top-k -> compacted sparse codes
  topk_select_kernel<<<B_DIM, blk, 0, stream>>>(acts, kest, idx, val, nnz,
                                                F_DIM);

  // Transpose W_dec once (overwrites bf16 weight region; GEMMs are done)
  const dim3 grid_tr(F_DIM / 32, D_DIM / 32);
  transpose_kernel<<<grid_tr, blk, 0, stream>>>(W_dec, WdT, D_DIM, F_DIM);

  // Sparse fp32 decode: out = enc @ W_dec^T + b_dec
  decoder_sparse_kernel<<<B_DIM, blk, 0, stream>>>(WdT, b_dec, idx, val, nnz,
                                                   out);
}
